// MS_Spatial_Bottleneck_Layer_83373905150242
// MI455X (gfx1250) — matrix operand order repaired and checked by reading, not compiled, not run on hardware
//
#include <hip/hip_runtime.h>

typedef __attribute__((ext_vector_type(16))) _Float16 v16h;
typedef __attribute__((ext_vector_type(8)))  _Float16 v8h;
typedef __attribute__((ext_vector_type(8)))  float    v8f;
typedef __attribute__((ext_vector_type(4)))  int      v4i;

#define EPSF 1e-5f
#define NB   64
#define CIN  256
#define PP   1600     /* T*V = 64*25 */
#define CMID 128
#define CB   32
#define KC   96       /* 3*CB */

// ---------- optional CDNA5 async global->LDS path ----------
#if defined(__has_builtin)
#if __has_builtin(__builtin_amdgcn_global_load_async_to_lds_b128)
#define HAVE_ASYNC_LDS 1
#endif
#endif
#ifndef HAVE_ASYNC_LDS
#define HAVE_ASYNC_LDS 0
#endif

#if HAVE_ASYNC_LDS
typedef __attribute__((address_space(1))) v4i glb_v4i;
typedef __attribute__((address_space(3))) v4i lds_v4i;
__device__ inline glb_v4i* glb_cast(const void* p) {
  return (glb_v4i*)(size_t)p;  // generic global: address identical
}
__device__ inline lds_v4i* lds_cast(void* p) {
  return (lds_v4i*)(uint32_t)(size_t)p;  // low 32 bits = LDS byte offset
}
__device__ inline void wait_async_all() {
#if __has_builtin(__builtin_amdgcn_s_wait_asynccnt)
  __builtin_amdgcn_s_wait_asynccnt(0);
#else
  asm volatile("s_wait_asynccnt 0" ::: "memory");
#endif
}
#endif

// ---------- WMMA fragment helpers (f16 16x16x32, wave32) ----------
// A tile 16(M)x32(K), row-major, row stride lda halves.
// lane<16: row M=lane, K0..7 -> halves0..7, K16..23 -> halves8..15;
// lane>=16: row M=lane-16, K8..15 / K24..31.
__device__ inline v16h load_a_frag(const _Float16* Abase, int lda, int lane) {
  int m    = lane & 15;
  int koff = (lane >> 4) * 8;
  const _Float16* row = Abase + (size_t)m * lda;
  v8h lo = *reinterpret_cast<const v8h*>(row + koff);
  v8h hi = *reinterpret_cast<const v8h*>(row + 16 + koff);
  v16h a;
#pragma unroll
  for (int j = 0; j < 8; ++j) { a[j] = lo[j]; a[8 + j] = hi[j]; }
  return a;
}

// B tile 32(K)x16(N) stored transposed as [N][K] (K contiguous), stride ldb halves.
// lane<16 -> N=lane, K0..7 / K16..23; lane>=16 -> N=lane-16, K8..15 / K24..31.
__device__ inline v16h load_b_frag(const _Float16* Bt, int ldb, int lane) {
  int n    = lane & 15;
  int koff = (lane >> 4) * 8;
  const _Float16* row = Bt + (size_t)n * ldb + koff;
  v8h lo = *reinterpret_cast<const v8h*>(row);
  v8h hi = *reinterpret_cast<const v8h*>(row + 16);
  v16h b;
#pragma unroll
  for (int j = 0; j < 8; ++j) { b[j] = lo[j]; b[8 + j] = hi[j]; }
  return b;
}

__device__ inline v8f wmma_f16(v16h a, v16h b, v8f c) {
  return __builtin_amdgcn_wmma_f32_16x16x32_f16(false, a, false, b, (short)0, c,
                                                false, false);
}

// ---------- prep: fold BN/bias, build A+PA, f16 weights ----------
__global__ __launch_bounds__(256) void k_prep(
    const float* __restrict__ A, const float* __restrict__ reduct_w,
    const float* __restrict__ reduct_b, const float* __restrict__ rg,
    const float* __restrict__ rb, const float* __restrict__ rm,
    const float* __restrict__ rv, const float* __restrict__ gcn_w,
    const float* __restrict__ gcn_b, const float* __restrict__ PA,
    const float* __restrict__ bg, const float* __restrict__ bb,
    const float* __restrict__ bm, const float* __restrict__ bv,
    const float* __restrict__ exp_w, const float* __restrict__ exp_b,
    const float* __restrict__ eg, const float* __restrict__ eb,
    const float* __restrict__ em, const float* __restrict__ ev,
    _Float16* __restrict__ Wr, _Float16* __restrict__ We,
    _Float16* __restrict__ Wg, float* __restrict__ ralpha,
    float* __restrict__ rbeta, float* __restrict__ ealpha,
    float* __restrict__ ebeta, float* __restrict__ Madj,
    float* __restrict__ salpha, float* __restrict__ sbeta) {
  int tid = threadIdx.x;
  for (int e = tid; e < CMID * CIN; e += 256) Wr[e] = (_Float16)reduct_w[e];
  for (int e = tid; e < CIN * CMID; e += 256) We[e] = (_Float16)exp_w[e];
  for (int e = tid; e < 4 * KC * CB; e += 256) Wg[e] = (_Float16)gcn_w[e];
  for (int e = tid; e < 4 * 1875; e += 256) Madj[e] = A[e % 1875] + PA[e];
  for (int e = tid; e < CMID; e += 256) {
    float s = rg[e] * rsqrtf(rv[e] + EPSF);
    ralpha[e] = s;
    rbeta[e]  = rb[e] - rm[e] * s + reduct_b[e] * s;
  }
  for (int e = tid; e < CIN; e += 256) {
    float s = eg[e] * rsqrtf(ev[e] + EPSF);
    ealpha[e] = s;
    ebeta[e]  = eb[e] - em[e] * s + exp_b[e] * s;
  }
  for (int e = tid; e < 4 * CB; e += 256)
    salpha[e] = bg[e] * rsqrtf(bv[e] + EPSF);
  // sbeta[i][c][w] = (sum_k gcn_b[i,k*32+c] * sum_v (A+PA)[i,k,v,w]) * scale + shift
  for (int e = tid; e < 4 * CB * 25; e += 256) {
    int i = e / 800, r = e % 800, c = r / 25, w = r % 25;
    float s     = bg[i * 32 + c] * rsqrtf(bv[i * 32 + c] + EPSF);
    float shift = bb[i * 32 + c] - bm[i * 32 + c] * s;
    float bias  = 0.f;
    for (int k = 0; k < 3; ++k) {
      float cs = 0.f;
      for (int v = 0; v < 25; ++v)
        cs += A[(k * 25 + v) * 25 + w] + PA[((i * 3 + k) * 25 + v) * 25 + w];
      bias += gcn_b[i * 96 + k * 32 + c] * cs;
    }
    sbeta[e] = bias * s + shift;
  }
}

// ---------- reduce conv: h[n][p][c] = relu(bn(Wr @ x)), f16, channel-contiguous ----------
__global__ __launch_bounds__(256) void k_reduce(
    const float* __restrict__ x, const _Float16* __restrict__ Wh,
    const float* __restrict__ alpha, const float* __restrict__ beta,
    _Float16* __restrict__ h) {
  __shared__ _Float16 xb[16 * 72];  // [p][k] for one K-chunk of 64, pad 72
  int ntile = blockIdx.x;           // 0..99
  int n     = blockIdx.y;
  int tid = threadIdx.x, lane = tid & 31, wave = tid >> 5;
  size_t xbase = (size_t)n * CIN * PP + (size_t)ntile * 16;
  int kl = tid >> 2, p4 = (tid & 3) * 4;  // 64 k-rows x 4 float4 cols
  v8f acc = {};
#pragma unroll
  for (int kc = 0; kc < CIN / 64; ++kc) {
    __syncthreads();
    {  // stage chunk [64k x 16p] -> xb[p][k], one float4 per thread
      float4 xv = *reinterpret_cast<const float4*>(
          &x[xbase + (size_t)(kc * 64 + kl) * PP + p4]);
      xb[(p4 + 0) * 72 + kl] = (_Float16)xv.x;
      xb[(p4 + 1) * 72 + kl] = (_Float16)xv.y;
      xb[(p4 + 2) * 72 + kl] = (_Float16)xv.z;
      xb[(p4 + 3) * 72 + kl] = (_Float16)xv.w;
    }
    if (kc + 1 < CIN / 64)
      __builtin_prefetch(&x[xbase + (size_t)((kc + 1) * 64 + kl) * PP + p4], 0, 1);
    __syncthreads();
#pragma unroll
    for (int s = 0; s < 2; ++s) {
      v16h a = load_a_frag(Wh + (size_t)(wave * 16) * CIN + kc * 64 + s * 32,
                           CIN, lane);
      v16h b = load_b_frag(xb + s * 32, 72, lane);
      acc = wmma_f16(a, b, acc);
    }
  }
  // epilogue: 8 consecutive channels per lane -> single b128 store
  int hi = lane >> 4, nn = lane & 15;
  int m0 = wave * 16 + hi * 8;
  v8h o;
#pragma unroll
  for (int r = 0; r < 8; ++r) {
    float v = acc[r] * alpha[m0 + r] + beta[m0 + r];
    o[r]    = (_Float16)(v > 0.f ? v : 0.f);
  }
  *reinterpret_cast<v8h*>(
      &h[((size_t)n * PP + (size_t)ntile * 16 + nn) * CMID + m0]) = o;
}

// ---------- fused stage: sp += chunk; g = Wg@sp; out = relu(bn(g x (A+PA))) ----------
// h layout [n][p][c]; overwrites channel chunk `stage` in place. Dynamic LDS.
__global__ __launch_bounds__(256) void k_stage(
    _Float16* __restrict__ h, const _Float16* __restrict__ Wgall,
    const float* __restrict__ Madj, const float* __restrict__ salpha,
    const float* __restrict__ sbeta, int stage) {
  extern __shared__ char smem[];
  _Float16* sp = (_Float16*)smem;                    // [400 p][40] = 32000 B
  _Float16* g  = (_Float16*)(smem + 32000);          // [400 p][96] = 76800 B
  _Float16* wg = (_Float16*)(smem + 108800);         //  6144 B
  float* M  = (float*)(smem + 114944);               //  7500 B
  float* sa = (float*)(smem + 122444);               //   128 B
  float* sb = sa + 32;                               //  3200 B (total 125772)
  int tb = blockIdx.x, n = blockIdx.y;
  int tid = threadIdx.x, lane = tid & 31, wave = tid >> 5;
  int    p0   = tb * 400;  // 16 t-rows * 25 v
  int    cb_i = stage * CB;
  size_t hn   = (size_t)n * PP;

  for (int e = tid; e < KC * CB; e += 256) wg[e] = Wgall[stage * KC * CB + e];
  for (int e = tid; e < 1875; e += 256) M[e] = Madj[stage * 1875 + e];
  if (tid < 32) sa[tid] = salpha[stage * 32 + tid];
  for (int e = tid; e < 800; e += 256) sb[e] = sbeta[stage * 800 + e];
  // spsum -> LDS [p][c], vectorized v8h
  for (int e = tid; e < 1600; e += 256) {
    int p = e >> 2, grp = (e & 3) * 8;
    size_t rowb = (hn + p0 + p) * CMID;
    v8h v = *reinterpret_cast<const v8h*>(&h[rowb + cb_i + grp]);
    if (stage > 0) {
      v8h u = *reinterpret_cast<const v8h*>(&h[rowb + cb_i - CB + grp]);
#pragma unroll
      for (int j = 0; j < 8; ++j) v[j] = (_Float16)((float)v[j] + (float)u[j]);
    }
    *reinterpret_cast<v8h*>(&sp[p * 40 + grp]) = v;
  }
  __syncthreads();
  // g[p][m] = (Wg(96x32) @ sp(32x400))^T: 6x25 WMMA tiles, K=32 in one shot
  for (int tile = wave; tile < 150; tile += 8) {
    int mt = tile / 25, nt = tile % 25;
    v16h a = load_a_frag(wg + mt * 16 * CB, CB, lane);
    v16h b = load_b_frag(sp + nt * 16 * 40, 40, lane);
    v8f acc = {};
    acc = wmma_f16(a, b, acc);
    int hi = lane >> 4, nn = lane & 15;
    v8h go;
#pragma unroll
    for (int r = 0; r < 8; ++r) go[r] = (_Float16)acc[r];
    *reinterpret_cast<v8h*>(&g[(nt * 16 + nn) * 96 + mt * 16 + hi * 8]) = go;
  }
  __syncthreads();
  // adjacency + folded bias/BN + relu; lanes sweep channels -> coalesced b16
  for (int e = tid; e < CB * 400; e += 256) {
    int pl = e >> 5, c = e & 31;       // pl = t*25 + w (output position)
    int t = pl / 25, w = pl % 25;
    float acc = 0.f;
#pragma unroll
    for (int k = 0; k < 3; ++k) {
      const _Float16* gr = g + (t * 25) * 96 + k * 32 + c;
      const float*    Mr = M + k * 625 + w;
#pragma unroll
      for (int v = 0; v < 25; ++v) acc += (float)gr[v * 96] * Mr[v * 25];
    }
    float val = acc * sa[c] + sb[c * 25 + w];
    val       = val > 0.f ? val : 0.f;
    h[(hn + p0 + pl) * CMID + cb_i + c] = (_Float16)val;
  }
}

// ---------- expand conv: out = relu(bn(We @ hcat) + x), f32 output ----------
__global__ __launch_bounds__(256) void k_expand(
    const _Float16* __restrict__ h, const _Float16* __restrict__ Wh,
    const float* __restrict__ alpha, const float* __restrict__ beta,
    const float* __restrict__ xres, float* __restrict__ out) {
  __shared__ _Float16 hb[16 * 72];  // [p][k] for one K-chunk of 64, pad 72
  int ntile = blockIdx.x, n = blockIdx.y;
  int tid = threadIdx.x, lane = tid & 31, wave = tid >> 5;
  int p0g = ntile * 16;
  v8f acc0 = {}, acc1 = {};
#pragma unroll
  for (int kc = 0; kc < CMID / 64; ++kc) {
    __syncthreads();
    if (tid < 128) {  // stage [16p][64c] tile: h is already [p][c]-contiguous
      int prow = tid >> 3, col8 = (tid & 7) * 8;
      size_t gidx = ((size_t)n * PP + p0g + prow) * CMID + kc * 64 + col8;
      _Float16* lp = hb + prow * 72 + col8;
#if HAVE_ASYNC_LDS
      __builtin_amdgcn_global_load_async_to_lds_b128(glb_cast(h + gidx),
                                                     lds_cast(lp), 0, 0);
#else
      *reinterpret_cast<v8h*>(lp) = *reinterpret_cast<const v8h*>(h + gidx);
#endif
    }
#if HAVE_ASYNC_LDS
    wait_async_all();
#endif
    __syncthreads();
#pragma unroll
    for (int s = 0; s < 2; ++s) {
      v16h b  = load_b_frag(hb + s * 32, 72, lane);
      v16h a0 = load_a_frag(Wh + (size_t)(wave * 16) * CMID + kc * 64 + s * 32,
                            CMID, lane);
      acc0 = wmma_f16(a0, b, acc0);
      v16h a1 = load_a_frag(
          Wh + (size_t)((wave + 8) * 16) * CMID + kc * 64 + s * 32, CMID, lane);
      acc1 = wmma_f16(a1, b, acc1);
    }
  }
  int hi = lane >> 4, nn = lane & 15;
  size_t pcol = (size_t)ntile * 16 + nn;
#pragma unroll
  for (int r = 0; r < 8; ++r) {
    int m0 = wave * 16 + r + hi * 8;
    int m1 = (wave + 8) * 16 + r + hi * 8;
    size_t i0 = ((size_t)n * CIN + m0) * PP + pcol;
    size_t i1 = ((size_t)n * CIN + m1) * PP + pcol;
    float v0 = acc0[r] * alpha[m0] + beta[m0] + xres[i0];
    float v1 = acc1[r] * alpha[m1] + beta[m1] + xres[i1];
    out[i0] = v0 > 0.f ? v0 : 0.f;
    out[i1] = v1 > 0.f ? v1 : 0.f;
  }
}

extern "C" void kernel_launch(void* const* d_in, const int* in_sizes, int n_in,
                              void* d_out, int out_size, void* d_ws,
                              size_t ws_size, hipStream_t stream) {
  const float* x        = (const float*)d_in[0];
  const float* A        = (const float*)d_in[1];
  const float* reduct_w = (const float*)d_in[2];
  const float* reduct_b = (const float*)d_in[3];
  const float* rg = (const float*)d_in[4];
  const float* rb = (const float*)d_in[5];
  const float* rm = (const float*)d_in[6];
  const float* rv = (const float*)d_in[7];
  const float* gcn_w = (const float*)d_in[8];
  const float* gcn_b = (const float*)d_in[9];
  const float* PA = (const float*)d_in[10];
  const float* bg = (const float*)d_in[11];
  const float* bb = (const float*)d_in[12];
  const float* bm = (const float*)d_in[13];
  const float* bv = (const float*)d_in[14];
  const float* exp_w = (const float*)d_in[15];
  const float* exp_b = (const float*)d_in[16];
  const float* eg = (const float*)d_in[17];
  const float* eb = (const float*)d_in[18];
  const float* em = (const float*)d_in[19];
  const float* ev = (const float*)d_in[20];

  char* ws = (char*)d_ws;
  size_t off = 0;
  auto carve = [&](size_t bytes) {
    void* p = ws + off;
    off += (bytes + 255) & ~(size_t)255;
    return p;
  };
  _Float16* Wr = (_Float16*)carve(CMID * CIN * 2);
  _Float16* We = (_Float16*)carve(CIN * CMID * 2);
  _Float16* Wg = (_Float16*)carve(4 * KC * CB * 2);
  float* ralpha = (float*)carve(CMID * 4);
  float* rbeta  = (float*)carve(CMID * 4);
  float* ealpha = (float*)carve(CIN * 4);
  float* ebeta  = (float*)carve(CIN * 4);
  float* Madj   = (float*)carve(4 * 1875 * 4);
  float* salpha = (float*)carve(4 * CB * 4);
  float* sbeta  = (float*)carve(4 * CB * 25 * 4);
  _Float16* hws = (_Float16*)carve((size_t)NB * CMID * PP * 2);
  (void)ws_size; (void)in_sizes; (void)n_in; (void)out_size;

  k_prep<<<1, 256, 0, stream>>>(A, reduct_w, reduct_b, rg, rb, rm, rv, gcn_w,
                                gcn_b, PA, bg, bb, bm, bv, exp_w, exp_b, eg, eb,
                                em, ev, Wr, We, Wg, ralpha, rbeta, ealpha,
                                ebeta, Madj, salpha, sbeta);
  k_reduce<<<dim3(100, NB), 256, 0, stream>>>(x, Wr, ralpha, rbeta, hws);
  const size_t STAGE_LDS = 125952;  // ~126 KB dynamic LDS (<= 320 KB/WGP)
  for (int i = 0; i < 4; ++i)
    k_stage<<<dim3(4, NB), 256, STAGE_LDS, stream>>>(hws, Wg, Madj, salpha,
                                                     sbeta, i);
  k_expand<<<dim3(100, NB), 256, 0, stream>>>(hws, We, ealpha, ebeta, x,
                                              (float*)d_out);
}
